// TransformerBlock_72559177498812
// MI455X (gfx1250) — matrix operand order repaired and checked by reading, not compiled, not run on hardware
//
#include <hip/hip_runtime.h>
#include <hip/hip_bf16.h>
#include <math.h>

// ---------- problem constants ----------
#define DIM      1024
#define N_HEADS  16
#define HEAD_DIM 64
#define FF_DIM   4096
#define BATCH    2
#define SEQ      2048
#define MTOT     (BATCH * SEQ)   // 4096 token rows
#define EPS      1e-8f

// ---------- vector types ----------
typedef _Float16 v16h __attribute__((ext_vector_type(16)));
typedef _Float16 v8h  __attribute__((ext_vector_type(8)));
typedef _Float16 v4h  __attribute__((ext_vector_type(4)));
typedef float    v8f  __attribute__((ext_vector_type(8)));
typedef unsigned u32x4 __attribute__((ext_vector_type(4)));
typedef unsigned u32x8 __attribute__((ext_vector_type(8)));

// ---------- WMMA fragment gather helpers (CDNA5 16-bit layouts) ----------
// A 16x32 (f16): lane L holds row M=L%16; halves = K in [g*8,g*8+8) and [16+g*8,+8), g=L/16.
// B 32x16 (f16): lane L holds col N=L%16; halves = K in [g*8,g*8+8) and [16+g*8,+8), g=L/16.
__device__ __forceinline__ v16h frag_from(const _Float16* p0, const _Float16* p1) {
    v8h lo = *(const v8h*)p0;
    v8h hi = *(const v8h*)p1;
    v16h r;
#pragma unroll
    for (int e = 0; e < 8; ++e) { r[e] = lo[e]; r[e + 8] = hi[e]; }
    return r;
}
__device__ __forceinline__ v16h load_frag_a(const _Float16* lds, int stride, int row, int kc, int g) {
    const _Float16* p = lds + row * stride + kc;
    return frag_from(p + g * 8, p + 16 + g * 8);
}
__device__ __forceinline__ v16h load_frag_b(const _Float16* lds, int stride, int row, int kc, int g) {
    const _Float16* p = lds + row * stride + kc;
    return frag_from(p + g * 8, p + 16 + g * 8);
}

__device__ __forceinline__ float red_max16(float v) {
    v = fmaxf(v, __shfl_xor(v, 1, 32));
    v = fmaxf(v, __shfl_xor(v, 2, 32));
    v = fmaxf(v, __shfl_xor(v, 4, 32));
    v = fmaxf(v, __shfl_xor(v, 8, 32));
    return v;
}
__device__ __forceinline__ float red_sum16(float v) {
    v += __shfl_xor(v, 1, 32);
    v += __shfl_xor(v, 2, 32);
    v += __shfl_xor(v, 4, 32);
    v += __shfl_xor(v, 8, 32);
    return v;
}

// ---------- Tensor Data Mover: 2D f16 tile (32 x 128 rows) -> LDS ----------
// D# group1 (uniform per kernel): data_size=2B, pad every 16 DWORDs by 4 DWORDs
// => LDS row stride 40 halves (== GSTR), matching the fragment-gather layout.
__device__ __forceinline__ u32x8 tdm_group1(int K, int N) {
    u32x8 g1;
    g1[0] = (1u << 16)        // data_size = 1 (2 bytes)
          | (1u << 20)        // pad_enable
          | (3u << 22)        // pad_interval: 2^(3+1) = 16 DWORDs
          | (3u << 25);       // pad_amount: 3+1 = 4 DWORDs
    g1[1] = ((unsigned)K & 0xFFFFu) << 16;                          // tensor_dim0[15:0]
    g1[2] = ((unsigned)K >> 16) | (((unsigned)N & 0xFFFFu) << 16);  // dim0[31:16] | dim1[15:0]
    g1[3] = ((unsigned)N >> 16) | (32u << 16);                      // dim1[31:16] | tile_dim0=32
    g1[4] = 128u;                                                   // tile_dim1=128, tile_dim2=0
    g1[5] = (unsigned)K;                                            // tensor_dim0_stride[31:0]
    g1[6] = 0u;                                                     // stride hi / dim1_stride lo
    g1[7] = 0u;
    return g1;
}
__device__ __forceinline__ void tdm_issue(unsigned lds_addr, unsigned long long gaddr, u32x8 g1) {
    u32x4 g0;
    g0[0] = 1u;                                    // count=1, user mode
    g0[1] = lds_addr;                              // lds_addr
    g0[2] = (unsigned)gaddr;                       // global_addr[31:0]
    g0[3] = (unsigned)(gaddr >> 32) | (2u << 30);  // global_addr[56:32] | type=2
    asm volatile("tensor_load_to_lds %0, %1" :: "s"(g0), "s"(g1) : "memory");
}

// =====================================================================
// Weight pre-pass: W[K][N] f32  ->  Wt[N][K] f16  (TDM-friendly, N-major)
// =====================================================================
__global__ void convert_transpose_w(const float* __restrict__ W, _Float16* __restrict__ Wt,
                                    int K, int N) {
    size_t idx = (size_t)blockIdx.x * blockDim.x + threadIdx.x;
    if (idx >= (size_t)K * N) return;
    int n = (int)(idx % N);
    int k = (int)(idx / N);
    Wt[(size_t)n * K + k] = (_Float16)W[idx];
}

// =====================================================================
// GEMM: C[M,N] = A[M,K] @ W[K,N] (+bias), A f32, W pre-converted f16 N-major.
// Block = 256 threads (8 waves). Tile 128x128, K-step 32.
// B tiles staged by TDM (double-buffered, issued by wave 0, TENSORcnt-pipelined).
// Each wave: 4(M) x 2(N) tiles of v_wmma_f32_16x16x32_f16.
// =====================================================================
#define BM 128
#define BN 128
#define BK 32
#define GSTR 40   // LDS half-stride: 16 DWORDs data + 4 DWORDs pad = 80B rows

__global__ __launch_bounds__(256)
void gemm_tdm_wmma(const float* __restrict__ A, const _Float16* __restrict__ Bt,
                   const float* __restrict__ bias, float* __restrict__ C,
                   int M, int N, int K) {
    __shared__ _Float16 As[BM * GSTR];
    __shared__ _Float16 Bs[2][BN * GSTR];

    const int tid  = threadIdx.x;
    const int lane = tid & 31;
    const int wave = tid >> 5;
    const int g    = lane >> 4;
    const int ln   = lane & 15;
    const int wm   = (wave & 1) * 64;
    const int wn   = (wave >> 1) * 32;
    const int rowBase = blockIdx.y * BM;
    const int colBase = blockIdx.x * BN;
    const int nsteps  = K / BK;

    const unsigned ldsB[2] = { (unsigned)(uintptr_t)&Bs[0][0],
                               (unsigned)(uintptr_t)&Bs[1][0] };
    const unsigned long long gB = (unsigned long long)(uintptr_t)
        (Bt + (size_t)colBase * K);
    const u32x8 g1 = tdm_group1(K, N);

    if (wave == 0) {
        tdm_issue(ldsB[0], gB, g1);                 // prime buffer 0 (kb = 0)
    }

    v8f acc[4][2] = {};

    for (int it = 0; it < nsteps; ++it) {
        const int kb = it * BK;

        // ---- stage A tile (128x32 f32 -> f16), loads batched before stores ----
        float4 va[4];
#pragma unroll
        for (int i = 0; i < 4; ++i) {
            int idx = tid + i * 256;
            int r   = idx >> 3;
            int c4  = (idx & 7) * 4;
            va[i] = *(const float4*)(A + (size_t)(rowBase + r) * K + kb + c4);
        }
#pragma unroll
        for (int i = 0; i < 4; ++i) {
            int idx = tid + i * 256;
            int r   = idx >> 3;
            int c4  = (idx & 7) * 4;
            v4h hv = { (_Float16)va[i].x, (_Float16)va[i].y,
                       (_Float16)va[i].z, (_Float16)va[i].w };
            *(v4h*)(&As[r * GSTR + c4]) = hv;
        }
        if (kb + BK < K)
            __builtin_prefetch(A + (size_t)(rowBase + (tid >> 3)) * K + kb + BK + (tid & 7) * 4, 0, 1);

        // ---- TDM pipeline: issue next B tile, wait for current ----
        if (wave == 0) {
            if (it + 1 < nsteps) {
                tdm_issue(ldsB[(it + 1) & 1], gB + (size_t)(kb + BK) * 2, g1);
                __builtin_amdgcn_s_wait_tensorcnt(1);   // current tile complete
            } else {
                __builtin_amdgcn_s_wait_tensorcnt(0);
            }
        }
        __syncthreads();

        const _Float16* Bcur = &Bs[it & 1][0];
        v16h af[4], bf[2];
#pragma unroll
        for (int mi = 0; mi < 4; ++mi)
            af[mi] = load_frag_a(As, GSTR, wm + mi * 16 + ln, 0, g);
#pragma unroll
        for (int ni = 0; ni < 2; ++ni)
            bf[ni] = load_frag_b(Bcur, GSTR, wn + ni * 16 + ln, 0, g);

#pragma unroll
        for (int mi = 0; mi < 4; ++mi)
#pragma unroll
            for (int ni = 0; ni < 2; ++ni)
                acc[mi][ni] = __builtin_amdgcn_wmma_f32_16x16x32_f16(
                    false, af[mi], false, bf[ni], (short)0, acc[mi][ni], false, false);
        __syncthreads();
    }

    // ---- epilogue: C layout (v8f): VGPR j -> M = j + 8*g, N = ln ----
#pragma unroll
    for (int mi = 0; mi < 4; ++mi)
#pragma unroll
        for (int ni = 0; ni < 2; ++ni) {
            int col = colBase + wn + ni * 16 + ln;
            float bv = bias ? bias[col] : 0.0f;
#pragma unroll
            for (int j = 0; j < 8; ++j) {
                int row = rowBase + wm + mi * 16 + 8 * g + j;
                C[(size_t)row * N + col] = acc[mi][ni][j] + bv;
            }
        }
}

// =====================================================================
// RoPE applied in place to q and k inside qkv[M][3][H][hd].
// =====================================================================
__global__ void rope_kernel(float* __restrict__ qkv) {
    int idx = blockIdx.x * blockDim.x + threadIdx.x;
    if (idx >= MTOT * N_HEADS * (HEAD_DIM / 2)) return;
    int i   = idx & 31;
    int h   = (idx >> 5) & (N_HEADS - 1);
    int row = idx >> 9;
    int s   = row & (SEQ - 1);
    float inv = __expf(-((float)(2 * i) / (float)HEAD_DIM) * 9.210340371976184f);
    float fr = (float)s * inv;
    float c = __cosf(fr), sn = __sinf(fr);
    size_t base = (size_t)row * (3 * DIM) + (size_t)h * HEAD_DIM;
    float* q = qkv + base;
    float* k = qkv + base + DIM;
    float q1 = q[i], q2 = q[i + 32];
    q[i]      = q1 * c - q2 * sn;
    q[i + 32] = q1 * sn + q2 * c;
    float k1 = k[i], k2 = k[i + 32];
    k[i]      = k1 * c - k2 * sn;
    k[i + 32] = k1 * sn + k2 * c;
}

// =====================================================================
// Flash attention, WMMA for S=Q·K^T and O+=P·V, online softmax.
// Block = 128 threads (4 waves), 64 query rows, key blocks of 64.
// =====================================================================
#define BQ   64
#define BKV  64
#define ASTR 72

__global__ __launch_bounds__(128)
void flash_attn_wmma(const float* __restrict__ qkv, float* __restrict__ out) {
    __shared__ _Float16 Qs[BQ * ASTR];
    __shared__ _Float16 Ks[BKV * ASTR];
    __shared__ _Float16 Vt[HEAD_DIM * ASTR];
    __shared__ _Float16 Ps[BQ * ASTR];

    const int tid  = threadIdx.x;
    const int lane = tid & 31;
    const int wave = tid >> 5;
    const int g    = lane >> 4;
    const int ln   = lane & 15;
    const int qb   = blockIdx.x * BQ;
    const int h    = blockIdx.y;
    const int b    = blockIdx.z;

    const float qscale = 0.125f * 1.44269504088896f;

#pragma unroll
    for (int i = 0; i < 8; ++i) {
        int idx = tid + i * 128;
        int r   = idx >> 4;
        int c4  = (idx & 15) * 4;
        size_t off = ((size_t)(b * SEQ + qb + r) * 3) * DIM + (size_t)h * HEAD_DIM + c4;
        float4 v = *(const float4*)(qkv + off);
        v4h hv = { (_Float16)(v.x * qscale), (_Float16)(v.y * qscale),
                   (_Float16)(v.z * qscale), (_Float16)(v.w * qscale) };
        *(v4h*)(&Qs[r * ASTR + c4]) = hv;
    }

    v8f o[4] = {};
    float mrow[8], lrow[8];
#pragma unroll
    for (int j = 0; j < 8; ++j) { mrow[j] = -1e30f; lrow[j] = 0.0f; }

    for (int kb = 0; kb < SEQ / BKV; ++kb) {
        __syncthreads();
#pragma unroll
        for (int i = 0; i < 8; ++i) {
            int idx = tid + i * 128;
            int r   = idx >> 4;
            int c4  = (idx & 15) * 4;
            size_t base = ((size_t)(b * SEQ + kb * BKV + r) * 3) * DIM + (size_t)h * HEAD_DIM + c4;
            float4 kv = *(const float4*)(qkv + base + DIM);
            float4 vv = *(const float4*)(qkv + base + 2 * DIM);
            v4h hk = { (_Float16)kv.x, (_Float16)kv.y, (_Float16)kv.z, (_Float16)kv.w };
            *(v4h*)(&Ks[r * ASTR + c4]) = hk;
            Vt[(c4 + 0) * ASTR + r] = (_Float16)vv.x;
            Vt[(c4 + 1) * ASTR + r] = (_Float16)vv.y;
            Vt[(c4 + 2) * ASTR + r] = (_Float16)vv.z;
            Vt[(c4 + 3) * ASTR + r] = (_Float16)vv.w;
        }
        __syncthreads();

        v16h aq[2];
        aq[0] = load_frag_a(Qs, ASTR, wave * 16 + ln, 0, g);
        aq[1] = load_frag_a(Qs, ASTR, wave * 16 + ln, 32, g);
        v8f s[4] = {};
#pragma unroll
        for (int nt = 0; nt < 4; ++nt) {
#pragma unroll
            for (int ks = 0; ks < 2; ++ks) {
                v16h bk = load_frag_b(Ks, ASTR, nt * 16 + ln, ks * 32, g);
                s[nt] = __builtin_amdgcn_wmma_f32_16x16x32_f16(
                    false, aq[ks], false, bk, (short)0, s[nt], false, false);
            }
        }

        float alpha[8];
#pragma unroll
        for (int j = 0; j < 8; ++j) {
            float v = fmaxf(fmaxf(s[0][j], s[1][j]), fmaxf(s[2][j], s[3][j]));
            v = red_max16(v);
            float mn = fmaxf(mrow[j], v);
            alpha[j] = exp2f(mrow[j] - mn);
            mrow[j]  = mn;
        }
#pragma unroll
        for (int j = 0; j < 8; ++j) {
            float rs = 0.0f;
#pragma unroll
            for (int nt = 0; nt < 4; ++nt) {
                float p = exp2f(s[nt][j] - mrow[j]);
                s[nt][j] = p;
                rs += p;
            }
            rs = red_sum16(rs);
            lrow[j] = lrow[j] * alpha[j] + rs;
#pragma unroll
            for (int dt = 0; dt < 4; ++dt) o[dt][j] *= alpha[j];
        }

#pragma unroll
        for (int nt = 0; nt < 4; ++nt)
#pragma unroll
            for (int j = 0; j < 8; ++j)
                Ps[(wave * 16 + j + 8 * g) * ASTR + nt * 16 + ln] = (_Float16)s[nt][j];
        asm volatile("s_wait_dscnt 0" ::: "memory");   // in-wave cross-lane LDS RAW

        v16h ap[2];
        ap[0] = load_frag_a(Ps, ASTR, wave * 16 + ln, 0, g);
        ap[1] = load_frag_a(Ps, ASTR, wave * 16 + ln, 32, g);
#pragma unroll
        for (int dt = 0; dt < 4; ++dt) {
#pragma unroll
            for (int ks = 0; ks < 2; ++ks) {
                v16h bv = load_frag_b(Vt, ASTR, dt * 16 + ln, ks * 32, g);
                o[dt] = __builtin_amdgcn_wmma_f32_16x16x32_f16(
                    false, ap[ks], false, bv, (short)0, o[dt], false, false);
            }
        }
    }

#pragma unroll
    for (int dt = 0; dt < 4; ++dt)
#pragma unroll
        for (int j = 0; j < 8; ++j) {
            int row = qb + wave * 16 + j + 8 * g;
            int col = h * HEAD_DIM + dt * 16 + ln;
            out[(size_t)(b * SEQ + row) * DIM + col] = o[dt][j] / lrow[j];
        }
}

// =====================================================================
// out[row] = scale * (resid+y) / (rms + eps), one 256-thread block per row
// =====================================================================
__global__ __launch_bounds__(256)
void add_rmsnorm_kernel(const float* __restrict__ resid, const float* __restrict__ y,
                        const float* __restrict__ scale, float* __restrict__ out) {
    __shared__ float red[8];
    const int row = blockIdx.x;
    const int tid = threadIdx.x;
    float v[4];
    float ss = 0.0f;
#pragma unroll
    for (int i = 0; i < 4; ++i) {
        int c = tid + i * 256;
        float t = resid[(size_t)row * DIM + c] + y[(size_t)row * DIM + c];
        v[i] = t;
        ss += t * t;
    }
#pragma unroll
    for (int m = 1; m < 32; m <<= 1) ss += __shfl_xor(ss, m, 32);
    if ((tid & 31) == 0) red[tid >> 5] = ss;
    __syncthreads();
    float total = 0.0f;
#pragma unroll
    for (int i = 0; i < 8; ++i) total += red[i];
    float rms = sqrtf(total) * 0.03125f;
    float inv = 1.0f / (rms + EPS);
#pragma unroll
    for (int i = 0; i < 4; ++i) {
        int c = tid + i * 256;
        out[(size_t)row * DIM + c] = scale[c] * v[i] * inv;
    }
}

__global__ void silu_gate_kernel(float* __restrict__ gbuf, const float* __restrict__ lbuf, size_t n) {
    size_t idx = (size_t)blockIdx.x * blockDim.x + threadIdx.x;
    if (idx >= n) return;
    float ge = gbuf[idx];
    gbuf[idx] = lbuf[idx] / (1.0f + __expf(-ge));
}

// =====================================================================
// launch
// =====================================================================
extern "C" void kernel_launch(void* const* d_in, const int* in_sizes, int n_in,
                              void* d_out, int out_size, void* d_ws, size_t ws_size,
                              hipStream_t stream) {
    const float* x          = (const float*)d_in[0];
    const float* w_qkv      = (const float*)d_in[1];
    const float* w_attn_out = (const float*)d_in[2];
    const float* w_gate     = (const float*)d_in[3];
    const float* b_gate     = (const float*)d_in[4];
    const float* w_lin      = (const float*)d_in[5];
    const float* b_lin      = (const float*)d_in[6];
    const float* w_ff_out   = (const float*)d_in[7];
    const float* scale1     = (const float*)d_in[8];
    const float* scale2     = (const float*)d_in[9];
    float* out = (float*)d_out;

    float* ws   = (float*)d_ws;
    float* qkv  = ws;                                  // 4096*3072
    float* attn = qkv  + (size_t)MTOT * 3 * DIM;       // 4096*1024
    float* proj = attn + (size_t)MTOT * DIM;           // 4096*1024
    float* x1   = proj + (size_t)MTOT * DIM;           // 4096*1024
    float* gbuf = x1   + (size_t)MTOT * DIM;           // 4096*4096
    float* lbuf = gbuf + (size_t)MTOT * FF_DIM;        // 4096*4096
    float* ff   = lbuf + (size_t)MTOT * FF_DIM;        // 4096*1024

    _Float16* hbase  = (_Float16*)(ff + (size_t)MTOT * DIM);
    _Float16* wqkv_h = hbase;                                  // [3072][1024]
    _Float16* watt_h = wqkv_h + (size_t)DIM * 3 * DIM;         // [1024][1024]
    _Float16* wgat_h = watt_h + (size_t)DIM * DIM;             // [4096][1024]
    _Float16* wlin_h = wgat_h + (size_t)DIM * FF_DIM;          // [4096][1024]
    _Float16* wffo_h = wlin_h + (size_t)DIM * FF_DIM;          // [1024][4096]

    dim3 blk(256);

    // 0) one-shot weight conversion to f16, N-major (TDM-friendly)
    {
        size_t n;
        n = (size_t)DIM * 3 * DIM;
        convert_transpose_w<<<(int)((n + 255) / 256), 256, 0, stream>>>(w_qkv, wqkv_h, DIM, 3 * DIM);
        n = (size_t)DIM * DIM;
        convert_transpose_w<<<(int)((n + 255) / 256), 256, 0, stream>>>(w_attn_out, watt_h, DIM, DIM);
        n = (size_t)DIM * FF_DIM;
        convert_transpose_w<<<(int)((n + 255) / 256), 256, 0, stream>>>(w_gate, wgat_h, DIM, FF_DIM);
        convert_transpose_w<<<(int)((n + 255) / 256), 256, 0, stream>>>(w_lin, wlin_h, DIM, FF_DIM);
        n = (size_t)FF_DIM * DIM;
        convert_transpose_w<<<(int)((n + 255) / 256), 256, 0, stream>>>(w_ff_out, wffo_h, FF_DIM, DIM);
    }

    // 1) qkv = x @ w_qkv
    gemm_tdm_wmma<<<dim3(3 * DIM / BN, MTOT / BM), blk, 0, stream>>>(
        x, wqkv_h, nullptr, qkv, MTOT, 3 * DIM, DIM);

    // 2) RoPE on q,k in place
    {
        int n = MTOT * N_HEADS * (HEAD_DIM / 2);
        rope_kernel<<<(n + 255) / 256, 256, 0, stream>>>(qkv);
    }

    // 3) attention -> attn[M][DIM]
    flash_attn_wmma<<<dim3(SEQ / BQ, N_HEADS, BATCH), dim3(128), 0, stream>>>(qkv, attn);

    // 4) proj = attn @ w_attn_out
    gemm_tdm_wmma<<<dim3(DIM / BN, MTOT / BM), blk, 0, stream>>>(
        attn, watt_h, nullptr, proj, MTOT, DIM, DIM);

    // 5) x1 = rmsnorm(x + proj) * scale1
    add_rmsnorm_kernel<<<MTOT, 256, 0, stream>>>(x, proj, scale1, x1);

    // 6) gate / lin GEMMs with fused bias
    gemm_tdm_wmma<<<dim3(FF_DIM / BN, MTOT / BM), blk, 0, stream>>>(
        x1, wgat_h, b_gate, gbuf, MTOT, FF_DIM, DIM);
    gemm_tdm_wmma<<<dim3(FF_DIM / BN, MTOT / BM), blk, 0, stream>>>(
        x1, wlin_h, b_lin, lbuf, MTOT, FF_DIM, DIM);

    // 7) h = sigmoid(g) * l  (into gbuf)
    {
        size_t n = (size_t)MTOT * FF_DIM;
        silu_gate_kernel<<<(int)((n + 255) / 256), 256, 0, stream>>>(gbuf, lbuf, n);
    }

    // 8) ff = h @ w_ff_out
    gemm_tdm_wmma<<<dim3(DIM / BN, MTOT / BM), blk, 0, stream>>>(
        gbuf, wffo_h, nullptr, ff, MTOT, DIM, FF_DIM);

    // 9) out = rmsnorm(x1 + ff) * scale2
    add_rmsnorm_kernel<<<MTOT, 256, 0, stream>>>(x1, ff, scale2, out);
}
